// MultiHeadAttention_35493609734354
// MI455X (gfx1250) — matrix operand order repaired and checked by reading, not compiled, not run on hardware
//
#include <hip/hip_runtime.h>

typedef __attribute__((ext_vector_type(16))) _Float16 v16h;
typedef __attribute__((ext_vector_type(8)))  float    v8f;

#define BATCH 4
#define SEQ   2048
#define DIM_  1024
#define NHEAD 16
#define HDIM  64
#define SCALE_ 0.125f   // HEAD_DIM^-0.5

// ---------------------------------------------------------------------------
// WMMA helpers (CDNA5 16x16x32 f16 -> f32)
// ---------------------------------------------------------------------------
__device__ __forceinline__ v8f wmma32(v16h a, v16h b, v8f c) {
  return __builtin_amdgcn_wmma_f32_16x16x32_f16(false, a, false, b, (short)0, c,
                                                false, false);
}

// A-matrix 16x32 f16 fragment (ISA 7.12.2): lanes 0-15 -> M=lane, K 0-7/16-23;
// lanes 16-31 -> M=lane-16, K 8-15/24-31. K-pairs are contiguous -> dword loads.
__device__ __forceinline__ v16h load_a_frag(const _Float16* base, int ldm,
                                            int m0, int koff) {
  const int l = threadIdx.x & 31;
  const _Float16* row = base + (size_t)(m0 + (l & 15)) * ldm + koff + ((l >> 4) << 3);
  v16h a;
#pragma unroll
  for (int j = 0; j < 8; ++j) {
    const int kk = ((j & 4) << 2) + ((j & 3) << 1);  // 0,2,4,6,16,18,20,22
    a[2 * j]     = row[kk];
    a[2 * j + 1] = row[kk + 1];
  }
  return a;
}

// B-matrix 32x16 f16 fragment from an N-major (K contiguous) buffer:
// lanes 0-15 -> N=n0+lane, K 0-7/16-23; lanes 16-31 -> N=n0+lane-16, K 8-15/24-31.
__device__ __forceinline__ v16h load_b_frag(const _Float16* baseT, int ldk,
                                            int n0, int koff) {
  const int l = threadIdx.x & 31;
  const _Float16* row = baseT + (size_t)(n0 + (l & 15)) * ldk + koff + ((l >> 4) << 3);
  v16h b;
#pragma unroll
  for (int j = 0; j < 16; ++j) b[j] = row[j + ((j >> 3) << 3)];
  return b;
}

// reductions across the 16-lane half-wave (C/D rows live across 16 lanes)
__device__ __forceinline__ float rmax16(float v) {
#pragma unroll
  for (int m = 1; m < 16; m <<= 1) v = fmaxf(v, __shfl_xor(v, m, 32));
  return v;
}
__device__ __forceinline__ float rsum16(float v) {
#pragma unroll
  for (int m = 1; m < 16; m <<= 1) v += __shfl_xor(v, m, 32);
  return v;
}

// ---------------------------------------------------------------------------
// Projection GEMM: C(M x N) = A(M x 1024, f32) @ W(1024 x N, f32), f16 out
// scattered into head-major Q / K / V buffers [b][h][n][d].
// Block: 256 thr (8 waves), tile 64x128; wave tile 32x32 (2x2 WMMA frags).
// K stepped by 64 (two 32-wide WMMA slices per stage): 8 WMMAs per barrier.
// ---------------------------------------------------------------------------
__global__ __launch_bounds__(256)
void proj_gemm_kernel(const float* __restrict__ A, const float* __restrict__ W,
                      int N, int mode, _Float16* __restrict__ qb,
                      _Float16* __restrict__ kb, _Float16* __restrict__ vb) {
  __shared__ _Float16 As[64][72];    // M x K slice (pad 8)
  __shared__ _Float16 WsT[128][72];  // N x K slice, transposed (pad 8)

  const int tid = threadIdx.x;
  const int wid = tid >> 5;
  const int l   = tid & 31;
  const int m_base = blockIdx.x * 64;
  const int n_base = blockIdx.y * 128;
  const int wm = (wid >> 2) * 32;
  const int wn = (wid & 3) * 32;

  v8f zero = {};
  v8f acc[2][2];
#pragma unroll
  for (int i = 0; i < 2; ++i)
#pragma unroll
    for (int j = 0; j < 2; ++j) acc[i][j] = zero;

  for (int k0 = 0; k0 < 1024; k0 += 64) {
    __syncthreads();
    {  // stage A tile 64x64, f32 -> f16
      const int row = tid >> 2;
      const int cg  = (tid & 3) * 16;
      const float* src = A + (size_t)(m_base + row) * 1024 + k0 + cg;
#pragma unroll
      for (int e = 0; e < 16; ++e) As[row][cg + e] = (_Float16)src[e];
      if (k0 + 64 < 1024) __builtin_prefetch(src + 64, 0, 3);
    }
#pragma unroll
    for (int e = 0; e < 32; ++e) {  // stage W tile 64x128 transposed
      const int idx = e * 256 + tid;
      const int kk = idx >> 7;
      const int nn = idx & 127;
      WsT[nn][kk] = (_Float16)W[(size_t)(k0 + kk) * N + (n_base + nn)];
    }
    if (k0 + 64 < 1024)
      __builtin_prefetch(W + (size_t)(k0 + 64 + (tid >> 3)) * N + n_base + (tid & 7) * 16, 0, 3);
    __syncthreads();

#pragma unroll
    for (int ks = 0; ks < 2; ++ks) {
      v16h a0 = load_a_frag(&As[0][0], 72, wm, ks * 32);
      v16h a1 = load_a_frag(&As[0][0], 72, wm + 16, ks * 32);
      v16h b0 = load_b_frag(&WsT[0][0], 72, wn, ks * 32);
      v16h b1 = load_b_frag(&WsT[0][0], 72, wn + 16, ks * 32);
      acc[0][0] = wmma32(a0, b0, acc[0][0]);
      acc[0][1] = wmma32(a0, b1, acc[0][1]);
      acc[1][0] = wmma32(a1, b0, acc[1][0]);
      acc[1][1] = wmma32(a1, b1, acc[1][1]);
    }
  }

  // epilogue: scatter into head-major layouts
#pragma unroll
  for (int mi = 0; mi < 2; ++mi)
#pragma unroll
    for (int ni = 0; ni < 2; ++ni)
#pragma unroll
      for (int rr = 0; rr < 8; ++rr) {
        const int gm = m_base + wm + mi * 16 + rr + ((l >> 4) << 3);
        const int gc = n_base + wn + ni * 16 + (l & 15);
        const _Float16 val = (_Float16)acc[mi][ni][rr];
        const int b    = gm >> 11;
        const int nrow = gm & 2047;
        if (mode == 0) {  // q: col = h*64+d
          const int h = gc >> 6, d = gc & 63;
          qb[(((size_t)(b * NHEAD + h)) * SEQ + nrow) * HDIM + d] = val;
        } else {          // kv: col = s*1024 + h*64 + d
          const int s = gc >> 10;
          const int cc = gc & 1023;
          const int h = cc >> 6, d = cc & 63;
          _Float16* dst = s ? vb : kb;
          dst[(((size_t)(b * NHEAD + h)) * SEQ + nrow) * HDIM + d] = val;
        }
      }
}

// ---------------------------------------------------------------------------
// Flash attention: block = 8 waves sharing one (b,h), 128 q rows; each wave
// owns 16 q rows. 32-key tiles staged in LDS: K rows via CDNA5 async
// memory->LDS copies (ASYNCcnt), V transposed via VGPR stores. 8 WMMAs per
// wave per tile. Online softmax with stats replicated per 16-lane half.
// ---------------------------------------------------------------------------
__global__ __launch_bounds__(256)
void attn_kernel(const _Float16* __restrict__ qb, const _Float16* __restrict__ kb,
                 const _Float16* __restrict__ vb, _Float16* __restrict__ rb) {
  __shared__ _Float16 Ks[32][72];       // [key][dim] (pad 8) -> Kt B frags
  __shared__ _Float16 VsT[64][40];      // [dim][key] (pad 8) -> V  B frags
  __shared__ _Float16 Ps[8][16][40];    // per-wave P transpose scratch

  const int tid = threadIdx.x;
  const int wid = tid >> 5;
  const int l   = tid & 31;
  const int qblk = blockIdx.x;
  const int h    = blockIdx.y;
  const int b    = blockIdx.z;
  const int bh   = b * NHEAD + h;
  const int q0   = qblk * 128 + wid * 16;

  const _Float16* qbase = qb + ((size_t)bh * SEQ + q0) * HDIM;
  const _Float16* kbase = kb + (size_t)bh * SEQ * HDIM;
  const _Float16* vbase = vb + (size_t)bh * SEQ * HDIM;

  const v16h qa0 = load_a_frag(qbase, HDIM, 0, 0);    // K(dim) 0..31
  const v16h qa1 = load_a_frag(qbase, HDIM, 0, 32);   // K(dim) 32..63

  v8f zero = {};
  v8f o[4];
#pragma unroll
  for (int i = 0; i < 4; ++i) o[i] = zero;
  float mstat[8], lstat[8];
#pragma unroll
  for (int r = 0; r < 8; ++r) { mstat[r] = -3.0e38f; lstat[r] = 0.0f; }

  const int key = tid >> 3;
  const int dg  = (tid & 7) * 8;
  const unsigned k_lds = (unsigned)(size_t)&Ks[key][dg];  // flat low 32b == LDS offset

  for (int kt = 0; kt < SEQ; kt += 32) {
    __syncthreads();  // previous tile fully consumed before overwrite
    {
      // K tile: async global->LDS (16B per lane), tracked by ASYNCcnt
      const _Float16* ksrc = kbase + (size_t)(kt + key) * HDIM + dg;
      asm volatile("global_load_async_to_lds_b128 %0, %1, off"
                   :: "v"(k_lds), "v"(ksrc) : "memory");
      // V tile: load + transposed store (async copy cannot transpose)
      const _Float16* vsrc = vbase + (size_t)(kt + key) * HDIM + dg;
#pragma unroll
      for (int e = 0; e < 8; ++e) VsT[dg + e][key] = vsrc[e];
      if (kt + 32 < SEQ) {
        __builtin_prefetch(ksrc + 32 * HDIM, 0, 3);
        __builtin_prefetch(vsrc + 32 * HDIM, 0, 3);
      }
    }
    asm volatile("s_wait_asynccnt 0x0" ::: "memory");
    __syncthreads();

    // S(16x32) = q(16x64) . K^T : two 16x16 C frags, 2 WMMAs each
    v8f s0 = zero, s1 = zero;
    s0 = wmma32(qa0, load_b_frag(&Ks[0][0], 72, 0, 0),   s0);
    s0 = wmma32(qa1, load_b_frag(&Ks[0][0], 72, 0, 32),  s0);
    s1 = wmma32(qa0, load_b_frag(&Ks[0][0], 72, 16, 0),  s1);
    s1 = wmma32(qa1, load_b_frag(&Ks[0][0], 72, 16, 32), s1);

    // online softmax across this 32-key tile
    float p0[8], p1[8];
#pragma unroll
    for (int r = 0; r < 8; ++r) {
      const float a = s0[r] * SCALE_;
      const float c = s1[r] * SCALE_;
      const float tmax  = rmax16(fmaxf(a, c));
      const float newm  = fmaxf(mstat[r], tmax);
      const float alpha = __expf(mstat[r] - newm);
      const float e0 = __expf(a - newm);
      const float e1 = __expf(c - newm);
      lstat[r] = lstat[r] * alpha + rsum16(e0 + e1);
      mstat[r] = newm;
      p0[r] = e0; p1[r] = e1;
#pragma unroll
      for (int nc = 0; nc < 4; ++nc) o[nc][r] *= alpha;
    }

    // C-layout P -> f16 A-layout via per-wave LDS scratch
#pragma unroll
    for (int r = 0; r < 8; ++r) {
      const int row = r + ((l >> 4) << 3);
      Ps[wid][row][(l & 15)]      = (_Float16)p0[r];
      Ps[wid][row][16 + (l & 15)] = (_Float16)p1[r];
    }
    asm volatile("s_wait_dscnt 0" ::: "memory");
    const v16h pa = load_a_frag(&Ps[wid][0][0], 40, 0, 0);
#pragma unroll
    for (int nc = 0; nc < 4; ++nc)
      o[nc] = wmma32(pa, load_b_frag(&VsT[0][0], 40, nc * 16, 0), o[nc]);
  }

  // finalize: O /= l, store f16 r[b][n][h*64+d]
#pragma unroll
  for (int r = 0; r < 8; ++r) {
    const float inv = 1.0f / lstat[r];
    const int row = q0 + r + ((l >> 4) << 3);
#pragma unroll
    for (int nc = 0; nc < 4; ++nc) {
      const int col = h * HDIM + nc * 16 + (l & 15);
      rb[((size_t)b * SEQ + row) * DIM_ + col] = (_Float16)(o[nc][r] * inv);
    }
  }
}

// ---------------------------------------------------------------------------
// Output projection: out(8192x1024 f32) = R(8192x1024 f16) @ Wo(1024x1024 f32)
// A frags come straight from global (row-major f16, contiguous K pairs);
// Wo tiles transposed through LDS. K stepped by 64.
// ---------------------------------------------------------------------------
__global__ __launch_bounds__(256)
void out_gemm_kernel(const _Float16* __restrict__ Rb, const float* __restrict__ Wo,
                     float* __restrict__ out) {
  __shared__ _Float16 WsT[128][72];

  const int tid = threadIdx.x;
  const int wid = tid >> 5;
  const int l   = tid & 31;
  const int m_base = blockIdx.x * 64;
  const int n_base = blockIdx.y * 128;
  const int wm = (wid >> 2) * 32;
  const int wn = (wid & 3) * 32;

  v8f zero = {};
  v8f acc[2][2];
#pragma unroll
  for (int i = 0; i < 2; ++i)
#pragma unroll
    for (int j = 0; j < 2; ++j) acc[i][j] = zero;

  for (int k0 = 0; k0 < DIM_; k0 += 64) {
    __syncthreads();
#pragma unroll
    for (int e = 0; e < 32; ++e) {
      const int idx = e * 256 + tid;
      const int kk = idx >> 7;
      const int nn = idx & 127;
      WsT[nn][kk] = (_Float16)Wo[(size_t)(k0 + kk) * DIM_ + (n_base + nn)];
    }
    if (k0 + 64 < DIM_)
      __builtin_prefetch(Wo + (size_t)(k0 + 64 + (tid >> 3)) * DIM_ + n_base + (tid & 7) * 16, 0, 3);
    __syncthreads();

    const _Float16* abase = Rb + (size_t)m_base * DIM_ + k0;
#pragma unroll
    for (int ks = 0; ks < 2; ++ks) {
      v16h a0 = load_a_frag(abase, DIM_, wm, ks * 32);
      v16h a1 = load_a_frag(abase, DIM_, wm + 16, ks * 32);
      v16h b0 = load_b_frag(&WsT[0][0], 72, wn, ks * 32);
      v16h b1 = load_b_frag(&WsT[0][0], 72, wn + 16, ks * 32);
      acc[0][0] = wmma32(a0, b0, acc[0][0]);
      acc[0][1] = wmma32(a0, b1, acc[0][1]);
      acc[1][0] = wmma32(a1, b0, acc[1][0]);
      acc[1][1] = wmma32(a1, b1, acc[1][1]);
    }
  }

#pragma unroll
  for (int mi = 0; mi < 2; ++mi)
#pragma unroll
    for (int ni = 0; ni < 2; ++ni)
#pragma unroll
      for (int rr = 0; rr < 8; ++rr) {
        const int gm = m_base + wm + mi * 16 + rr + ((l >> 4) << 3);
        const int gc = n_base + wn + ni * 16 + (l & 15);
        out[(size_t)gm * DIM_ + gc] = acc[mi][ni][rr];
      }
}

// ---------------------------------------------------------------------------
extern "C" void kernel_launch(void* const* d_in, const int* in_sizes, int n_in,
                              void* d_out, int out_size, void* d_ws, size_t ws_size,
                              hipStream_t stream) {
  (void)in_sizes; (void)n_in; (void)out_size; (void)ws_size;
  const float* x   = (const float*)d_in[0];
  const float* y   = (const float*)d_in[1];
  const float* Wq  = (const float*)d_in[2];
  const float* Wkv = (const float*)d_in[3];
  const float* Wo  = (const float*)d_in[4];
  float* out = (float*)d_out;

  const size_t elems = (size_t)BATCH * NHEAD * SEQ * HDIM;  // 8M f16 each
  _Float16* qb = (_Float16*)d_ws;
  _Float16* kb = qb + elems;
  _Float16* vb = kb + elems;
  _Float16* rb = vb + elems;   // [B*SEQ][1024] f16, total ws use = 64 MB

  proj_gemm_kernel<<<dim3(128, 8),  256, 0, stream>>>(x, Wq, 1024, 0, qb, kb, vb);
  proj_gemm_kernel<<<dim3(128, 16), 256, 0, stream>>>(y, Wkv, 2048, 1, qb, kb, vb);
  attn_kernel<<<dim3(16, NHEAD, BATCH), 256, 0, stream>>>(qb, kb, vb, rb);
  out_gemm_kernel<<<dim3(128, 8), 256, 0, stream>>>(rb, Wo, out);
}
